// CapsNet_71880572665971
// MI455X (gfx1250) — matrix operand order repaired and checked by reading, not compiled, not run on hardware
//
#include <hip/hip_runtime.h>
#include <hip/hip_bf16.h>

// ---------------------------------------------------------------------------
// CapsNet forward for MI455X (gfx1250, wave32).
// Dominant cost: [512,102400]x[102400,768] GEMM -> bf16 WMMA, split-K=16.
// Pipeline per K-step: issue global loads (regs) -> 16 WMMAs/wave -> convert+LDS
// store -> barrier, so global latency hides under the WMMA burst.
// log-normalization folded to one v_log_f32 + one FMA per element.
// Deterministic staged reductions everywhere (no float atomics).
// ---------------------------------------------------------------------------

typedef __attribute__((ext_vector_type(8)))  float          v8f;
typedef __attribute__((ext_vector_type(4)))  float          v4f;
typedef __attribute__((ext_vector_type(16))) __bf16         v16bf;
typedef __attribute__((ext_vector_type(4)))  unsigned int   v4u;
typedef __attribute__((ext_vector_type(8)))  unsigned int   v8u;
typedef __attribute__((ext_vector_type(2)))  unsigned int   v2u;

#define B_SZ     512
#define EMB      768
#define DCT_K    102400LL
#define ROUTES   192
#define NCAPS    2
#define OUTCH    64
#define SPLITS   16
#define KSTEP    64
#define LDS_LD   72        // bf16 row stride (144 B, 16-B aligned)
#define LN2F     0.6931471805599453f

__device__ __forceinline__ unsigned short f2bf(float f) {
  unsigned u = __builtin_bit_cast(unsigned, f);
  u += 0x7FFFu + ((u >> 16) & 1u);          // round-to-nearest-even
  return (unsigned short)(u >> 16);
}

// fast ln(|x|+1e-12): argument is >= 1e-12 (never denormal/zero/negative),
// so hardware v_log_f32 is safe and ~1 ulp in log2.
__device__ __forceinline__ float fast_logabs(float x) {
  return __log2f(fabsf(x) + 1e-12f) * LN2F;
}

// A fragment 16x32 bf16 (8 VGPRs): lane L -> row M=L&15, h=L>>4;
// elements 0..7 = K[ks+8h .. +7], elements 8..15 = K[ks+16+8h .. +7].
__device__ __forceinline__ v16bf load_frag_a(const unsigned short* row0, int lane, int ks) {
  int r = lane & 15, h = lane >> 4;
  const char* p = (const char*)(row0 + (size_t)r * LDS_LD) + 2 * ks;
  v4u lo = *(const v4u*)(p + 16 * h);
  v4u hi = *(const v4u*)(p + 32 + 16 * h);
  v8u regs = { lo[0], lo[1], lo[2], lo[3], hi[0], hi[1], hi[2], hi[3] };
  return __builtin_bit_cast(v16bf, regs);
}

// B fragment 32x16 bf16 (8 VGPRs): lane L -> col N=L&15, h=L>>4;
// elements 0..7 = K[ks+8h .. +7], elements 8..15 = K[ks+16+8h .. +7].
__device__ __forceinline__ v16bf load_frag_b(const unsigned short* row0, int lane, int ks) {
  int r = lane & 15, h = lane >> 4;
  const char* p = (const char*)(row0 + (size_t)r * LDS_LD) + 2 * ks;
  v4u lo = *(const v4u*)(p + 16 * h);
  v4u hi = *(const v4u*)(p + 32 + 16 * h);
  v8u regs = { lo[0], lo[1], lo[2], lo[3], hi[0], hi[1], hi[2], hi[3] };
  return __builtin_bit_cast(v16bf, regs);
}

// ---------------------------------------------------------------------------
// Pass 1/2: deterministic global mean / unbiased-std of log(|DCT|+1e-12)
// (same fast-log transform as the GEMM staging, so stats are self-consistent)
// ---------------------------------------------------------------------------
__global__ __launch_bounds__(256) void stats_pass1(const float* __restrict__ x,
                                                   long long n, float* __restrict__ part) {
  __shared__ float s1[256], s2[256];
  int t = threadIdx.x;
  float a = 0.f, b = 0.f;
  for (long long i = (long long)blockIdx.x * 256 + t; i < n;
       i += (long long)gridDim.x * 256) {
    float v = fast_logabs(x[i]);
    a += v; b += v * v;
  }
  s1[t] = a; s2[t] = b; __syncthreads();
  for (int s = 128; s > 0; s >>= 1) {
    if (t < s) { s1[t] += s1[t + s]; s2[t] += s2[t + s]; }
    __syncthreads();
  }
  if (t == 0) { part[blockIdx.x] = s1[0]; part[gridDim.x + blockIdx.x] = s2[0]; }
}

__global__ __launch_bounds__(256) void stats_pass2(const float* __restrict__ part, int nb,
                                                   long long n, float* __restrict__ stats) {
  __shared__ double d1[256], d2[256];
  int t = threadIdx.x;
  double a = 0.0, b = 0.0;
  for (int i = t; i < nb; i += 256) { a += (double)part[i]; b += (double)part[nb + i]; }
  d1[t] = a; d2[t] = b; __syncthreads();
  for (int s = 128; s > 0; s >>= 1) {
    if (t < s) { d1[t] += d1[t + s]; d2[t] += d2[t + s]; }
    __syncthreads();
  }
  if (t == 0) {
    double sum = d1[0], sq = d2[0];
    double mean = sum / (double)n;
    double var  = (sq - sum * sum / (double)n) / (double)(n - 1);  // ddof=1
    double istd = 1.0 / sqrt(var);
    // folded affine transform: a' = fma(log2(|a|+eps), ln2*istd, -mean*istd)
    stats[0] = (float)(-mean * istd);        // c0
    stats[1] = (float)((double)LN2F * istd); // c1
  }
}

// ---------------------------------------------------------------------------
// bf16 WMMA NT-GEMM:  Cp[z][m][n] = sum_{k in chunk z} A'[m,k] * Bm[n,k]
// MODE==1: A' = fma(log2(|A|+1e-12), c1, c0)   (folded into staging)
// Block tile 128x128, 256 threads (8 waves), wave tile 32x64 (2x4 WMMA frags),
// K-step 64, double-buffered LDS, load->compute->convert pipeline.
// ---------------------------------------------------------------------------
struct TileRegs { v4f a[8]; v4f b[8]; };

__device__ __forceinline__ void tile_load(TileRegs& tr,
                                          const float* __restrict__ A,
                                          const float* __restrict__ Bm,
                                          long long ldk, int bm, int bn, long long k0,
                                          int tid) {
  const int cr = tid >> 4;          // 0..15
  const int cc = (tid & 15) << 2;   // 0,4,..,60
#pragma unroll
  for (int i = 0; i < 8; ++i) {
    const float* sa = A + (size_t)(bm + cr + (i << 4)) * ldk + k0 + cc;
    tr.a[i] = *(const v4f*)sa;
    __builtin_prefetch(sa + KSTEP, 0, 0);      // global_prefetch_b8, next K step
  }
#pragma unroll
  for (int i = 0; i < 8; ++i) {
    const float* sb = Bm + (size_t)(bn + cr + (i << 4)) * ldk + k0 + cc;
    tr.b[i] = *(const v4f*)sb;
    __builtin_prefetch(sb + KSTEP, 0, 0);
  }
}

template <int MODE>
__device__ __forceinline__ void tile_store(const TileRegs& tr,
                                           unsigned short* __restrict__ dstA,
                                           unsigned short* __restrict__ dstB,
                                           int tid, float c0, float c1) {
  const int cr = tid >> 4;
  const int cc = (tid & 15) << 2;
#pragma unroll
  for (int i = 0; i < 8; ++i) {
    int r = cr + (i << 4);
    float f0 = tr.a[i][0], f1 = tr.a[i][1], f2 = tr.a[i][2], f3 = tr.a[i][3];
    if (MODE) {
      f0 = fmaf(__log2f(fabsf(f0) + 1e-12f), c1, c0);
      f1 = fmaf(__log2f(fabsf(f1) + 1e-12f), c1, c0);
      f2 = fmaf(__log2f(fabsf(f2) + 1e-12f), c1, c0);
      f3 = fmaf(__log2f(fabsf(f3) + 1e-12f), c1, c0);
    }
    unsigned p0 = (unsigned)f2bf(f0) | ((unsigned)f2bf(f1) << 16);
    unsigned p1 = (unsigned)f2bf(f2) | ((unsigned)f2bf(f3) << 16);
    *(v2u*)(&dstA[r * LDS_LD + cc]) = (v2u){p0, p1};
  }
#pragma unroll
  for (int i = 0; i < 8; ++i) {
    int r = cr + (i << 4);
    unsigned p0 = (unsigned)f2bf(tr.b[i][0]) | ((unsigned)f2bf(tr.b[i][1]) << 16);
    unsigned p1 = (unsigned)f2bf(tr.b[i][2]) | ((unsigned)f2bf(tr.b[i][3]) << 16);
    *(v2u*)(&dstB[r * LDS_LD + cc]) = (v2u){p0, p1};
  }
}

template <int MODE>
__global__ __launch_bounds__(256) void wmma_gemm_nt(
    const float* __restrict__ A, const float* __restrict__ Bm,
    float* __restrict__ Cp, int M, int N, long long ldk,
    long long kChunk, const float* __restrict__ stats) {
  __shared__ alignas(16) unsigned short Abuf[2][128 * LDS_LD];
  __shared__ alignas(16) unsigned short Bbuf[2][128 * LDS_LD];

  const int tid  = threadIdx.x;
  const int lane = tid & 31;
  const int wid  = tid >> 5;
  const int wm   = (wid & 3) * 32;   // wave M offset in block tile
  const int wn   = (wid >> 2) * 64;  // wave N offset in block tile
  const int bm   = blockIdx.y * 128;
  const int bn   = blockIdx.x * 128;

  float c0 = 0.f, c1 = 1.f;
  if (MODE) { c0 = stats[0]; c1 = stats[1]; }

  const long long kBegin = (long long)blockIdx.z * kChunk;
  const int nsteps = (int)(kChunk / KSTEP);

  v8f acc[2][4] = {};
  TileRegs tr;

  // prologue: stage first tile into buffer 0
  tile_load(tr, A, Bm, ldk, bm, bn, kBegin, tid);
  tile_store<MODE>(tr, Abuf[0], Bbuf[0], tid, c0, c1);
  __syncthreads();

  for (int s = 0; s < nsteps; ++s) {
    const int cur = s & 1;
    const bool hasNext = (s + 1 < nsteps);

    // 1) issue next tile's global loads (kept in regs, no wait yet)
    if (hasNext)
      tile_load(tr, A, Bm, ldk, bm, bn, kBegin + (long long)(s + 1) * KSTEP, tid);

    // 2) compute current tile from LDS while loads are in flight
#pragma unroll
    for (int ks = 0; ks < KSTEP; ks += 32) {
      v16bf af[2];
      af[0] = load_frag_a(&Abuf[cur][(wm + 0) * LDS_LD], lane, ks);
      af[1] = load_frag_a(&Abuf[cur][(wm + 16) * LDS_LD], lane, ks);
      v16bf bf_[4];
#pragma unroll
      for (int j = 0; j < 4; ++j)
        bf_[j] = load_frag_b(&Bbuf[cur][(wn + j * 16) * LDS_LD], lane, ks);
#pragma unroll
      for (int i = 0; i < 2; ++i)
#pragma unroll
        for (int j = 0; j < 4; ++j)
          acc[i][j] = __builtin_amdgcn_wmma_f32_16x16x32_bf16(
              false, af[i], false, bf_[j], (short)0, acc[i][j], false, false);
    }

    // 3) convert + store next tile to the other LDS buffer
    if (hasNext)
      tile_store<MODE>(tr, Abuf[cur ^ 1], Bbuf[cur ^ 1], tid, c0, c1);
    __syncthreads();
  }

  // ---- write partial C for this split ----
  float* Cz = Cp + (size_t)blockIdx.z * (size_t)M * (size_t)N;
  const int mo  = 8 * (lane >> 4);
  const int col = bn + wn + (lane & 15);
#pragma unroll
  for (int i = 0; i < 2; ++i)
#pragma unroll
    for (int j = 0; j < 4; ++j) {
      int rbase = bm + wm + i * 16 + mo;
      int cbase = col + j * 16;
#pragma unroll
      for (int q = 0; q < 8; ++q)
        Cz[(size_t)(rbase + q) * N + cbase] = acc[i][j][q];
    }
}

// ---------------------------------------------------------------------------
// Epilogues / small kernels
// ---------------------------------------------------------------------------
__global__ __launch_bounds__(256) void dct_epilogue(const float* __restrict__ Cp,
                                                    const float* __restrict__ bias,
                                                    float* __restrict__ out) {
  int i = blockIdx.x * 256 + threadIdx.x;
  if (i >= B_SZ * EMB) return;
  int e = i % EMB;
  float s = bias[e];
#pragma unroll
  for (int z = 0; z < SPLITS; ++z) s += Cp[(size_t)z * B_SZ * EMB + i];
  out[i] = fmaxf(s, 0.f);
}

// u[b, r=64m+o, k] = squash_k( Cp_prim[m][b][k*64+o] + bias_m[k*64+o] )
__global__ __launch_bounds__(256) void prim_epilogue(const float* __restrict__ Cp,
                                                     const float* __restrict__ b0,
                                                     const float* __restrict__ b1,
                                                     const float* __restrict__ b2,
                                                     float* __restrict__ u) {
  int idx = blockIdx.x * 256 + threadIdx.x;        // b*192 + r
  if (idx >= B_SZ * ROUTES) return;
  int b = idx / ROUTES, r = idx % ROUTES;
  int m = r / 64, o = r % 64;
  const float* C    = Cp + (size_t)m * B_SZ * 512 + (size_t)b * 512;
  const float* bias = (m == 0) ? b0 : ((m == 1) ? b1 : b2);
  float x[8]; float sq = 0.f;
#pragma unroll
  for (int k = 0; k < 8; ++k) {
    float t = C[k * 64 + o] + bias[k * 64 + o];
    x[k] = t; sq += t * t;
  }
  float scale = sq / ((1.f + sq) * sqrtf(sq + 1e-7f));
#pragma unroll
  for (int k = 0; k < 8; ++k) u[(size_t)idx * 8 + k] = x[k] * scale;
}

__global__ void zero_bij(float* b) { if (threadIdx.x < ROUTES * NCAPS) b[threadIdx.x] = 0.f; }

// softmax over routes (axis 0) for each capsule column c; 1 block per c.
__global__ void softmax_routes(const float* __restrict__ bij, float* __restrict__ cij) {
  __shared__ float sh[ROUTES];
  __shared__ float mx, sm;
  int c = blockIdx.x, r = threadIdx.x;
  float x = bij[r * NCAPS + c];
  sh[r] = x; __syncthreads();
  if (r == 0) { float m = sh[0]; for (int i = 1; i < ROUTES; ++i) m = fmaxf(m, sh[i]); mx = m; }
  __syncthreads();
  float e = expf(x - mx);
  sh[r] = e; __syncthreads();
  if (r == 0) { float s = 0.f; for (int i = 0; i < ROUTES; ++i) s += sh[i]; sm = s; }
  __syncthreads();
  cij[r * NCAPS + c] = e / sm;
}

// s[b,c,o] = sum_r c[r,c] * (W_digit[r,c,o,:] . u[b,r,:]) ; v = elemwise squash
__global__ __launch_bounds__(128) void routing_sv(const float* __restrict__ u,
                                                  const float* __restrict__ Wd,
                                                  const float* __restrict__ cij,
                                                  float* __restrict__ vout) {
  __shared__ float ush[ROUTES * 8];
  __shared__ float csh[ROUTES * NCAPS];
  int b = blockIdx.x, t = threadIdx.x;          // t: c*64 + o
  int c = t >> 6, o = t & 63;
  for (int i = t; i < ROUTES * 8; i += 128) ush[i] = u[(size_t)b * ROUTES * 8 + i];
  for (int i = t; i < ROUTES * NCAPS; i += 128) csh[i] = cij[i];
  __syncthreads();
  float s = 0.f;
  for (int r = 0; r < ROUTES; ++r) {
    const float* w = Wd + ((size_t)(r * NCAPS + c) * 64 + o) * 8;
    float uh = 0.f;
#pragma unroll
    for (int i = 0; i < 8; ++i) uh += w[i] * ush[r * 8 + i];
    s += csh[r * NCAPS + c] * uh;
  }
  float sq = s * s;
  vout[(size_t)b * 128 + t] = sq * s / ((1.f + sq) * sqrtf(sq + 1e-7f));
}

// b_ij[r,c] += mean_b sum_o u_hat[b,r,c,o]*v[b,c,o]; deterministic fixed tree.
__global__ __launch_bounds__(256) void routing_agree(const float* __restrict__ u,
                                                     const float* __restrict__ Wd,
                                                     const float* __restrict__ v,
                                                     float* __restrict__ bij) {
  __shared__ float wsh[512];
  __shared__ float red[256];
  int rc = blockIdx.x;                       // r*2 + c
  int c = rc & 1;
  int t = threadIdx.x;
  for (int i = t; i < 512; i += 256) wsh[i] = Wd[(size_t)rc * 512 + i];
  __syncthreads();
  float acc = 0.f;
  for (int b = t; b < B_SZ; b += 256) {
    const float* vb = v + (size_t)b * 128 + c * 64;
    float g[8] = {0, 0, 0, 0, 0, 0, 0, 0};
    for (int o = 0; o < 64; ++o) {
      float vv = vb[o];
#pragma unroll
      for (int i = 0; i < 8; ++i) g[i] += wsh[o * 8 + i] * vv;
    }
    const float* ub = u + ((size_t)b * ROUTES + (rc >> 1)) * 8;
    float a = 0.f;
#pragma unroll
    for (int i = 0; i < 8; ++i) a += ub[i] * g[i];
    acc += a;
  }
  red[t] = acc; __syncthreads();
  for (int s = 128; s > 0; s >>= 1) {
    if (t < s) red[t] += red[t + s];
    __syncthreads();
  }
  if (t == 0) bij[rc] += red[0] / (float)B_SZ;
}

// ---------------------------------------------------------------------------
extern "C" void kernel_launch(void* const* d_in, const int* in_sizes, int n_in,
                              void* d_out, int out_size, void* d_ws, size_t ws_size,
                              hipStream_t stream) {
  const float* img_emb  = (const float*)d_in[0];
  const float* capt_emb = (const float*)d_in[1];
  const float* dct      = (const float*)d_in[2];
  const float* W_emb    = (const float*)d_in[3];
  const float* b_emb    = (const float*)d_in[4];
  const float* W_img    = (const float*)d_in[5];
  const float* b_img    = (const float*)d_in[6];
  const float* W_capt   = (const float*)d_in[7];
  const float* b_capt   = (const float*)d_in[8];
  const float* W_dct    = (const float*)d_in[9];
  const float* b_dct    = (const float*)d_in[10];
  const float* W_digit  = (const float*)d_in[11];
  float* out = (float*)d_out;

  // workspace layout (floats), ~33 MB total
  float* ws      = (float*)d_ws;
  float* part    = ws;                                   // 2048
  float* stats   = ws + 2048;                            // 2
  float* Cp_big  = ws + 4096;                            // 16*512*768
  float* dct_emb = Cp_big + (size_t)SPLITS * B_SZ * EMB; // 512*768
  float* Cp_prim = dct_emb + (size_t)B_SZ * EMB;         // 3*512*512
  float* u       = Cp_prim + 3ull * B_SZ * 512;          // 512*192*8
  float* vbuf    = u + (size_t)B_SZ * ROUTES * 8;        // 512*128
  float* bij     = vbuf + (size_t)B_SZ * 128;            // 384
  float* cij     = bij + ROUTES * NCAPS;                 // 384

  const long long nDCT = (long long)B_SZ * DCT_K;

  // 1) global log-stats (two-stage deterministic reduction)
  stats_pass1<<<1024, 256, 0, stream>>>(dct, nDCT, part);
  stats_pass2<<<1, 256, 0, stream>>>(part, 1024, nDCT, stats);

  // 2) main GEMM: dct_emb_raw = norm(log|DCT|) @ W_emb^T, split-K=16
  {
    dim3 grid(EMB / 128, B_SZ / 128, SPLITS);
    wmma_gemm_nt<1><<<grid, 256, 0, stream>>>(dct, W_emb, Cp_big, B_SZ, EMB,
                                              DCT_K, DCT_K / SPLITS, stats);
  }
  dct_epilogue<<<(B_SZ * EMB + 255) / 256, 256, 0, stream>>>(Cp_big, b_emb, dct_emb);

  // 3) PrimaryCaps GEMMs: [512,768] @ [512,768]^T per modality
  {
    dim3 grid(512 / 128, B_SZ / 128, 1);
    wmma_gemm_nt<0><<<grid, 256, 0, stream>>>(img_emb, W_img,
                                              Cp_prim + 0ull * B_SZ * 512,
                                              B_SZ, 512, EMB, EMB, nullptr);
    wmma_gemm_nt<0><<<grid, 256, 0, stream>>>(capt_emb, W_capt,
                                              Cp_prim + 1ull * B_SZ * 512,
                                              B_SZ, 512, EMB, EMB, nullptr);
    wmma_gemm_nt<0><<<grid, 256, 0, stream>>>(dct_emb, W_dct,
                                              Cp_prim + 2ull * B_SZ * 512,
                                              B_SZ, 512, EMB, EMB, nullptr);
  }
  prim_epilogue<<<(B_SZ * ROUTES + 255) / 256, 256, 0, stream>>>(Cp_prim, b_img,
                                                                 b_capt, b_dct, u);

  // 4) dynamic routing, 3 iterations (u_hat recomputed on the fly; W_digit is L2-resident)
  zero_bij<<<1, 384, 0, stream>>>(bij);
  for (int it = 0; it < 3; ++it) {
    softmax_routes<<<NCAPS, ROUTES, 0, stream>>>(bij, cij);
    routing_sv<<<B_SZ, 128, 0, stream>>>(u, W_digit, cij, (it == 2) ? out : vbuf);
    if (it < 2)
      routing_agree<<<ROUTES * NCAPS, 256, 0, stream>>>(u, W_digit, vbuf, bij);
  }

  (void)in_sizes; (void)n_in; (void)out_size; (void)ws_size;
}
